// FastDecoderBlock_19894288515516
// MI455X (gfx1250) — matrix operand order repaired and checked by reading, not compiled, not run on hardware
//
#include <hip/hip_runtime.h>
#include <stdint.h>

#define B_    2
#define S_    2048
#define D_    1024
#define H_    16
#define HD_   64
#define DFF_  4096
#define MROWS (B_*S_)   // 4096

typedef __attribute__((ext_vector_type(16))) __bf16 v16bf;
typedef __attribute__((ext_vector_type(8)))  float  v8f;

union Frag { unsigned int u[8]; v16bf v; };

__device__ __forceinline__ unsigned short f2bf(float f) {
  unsigned int u = __float_as_uint(f);
  u = u + 0x7FFFu + ((u >> 16) & 1u);   // round-to-nearest-even
  return (unsigned short)(u >> 16);
}

// K offset inside a 16x32 bf16 A-fragment for VGPR v, lane-half hsel
// (contiguous order, matching B fragments: K = 16*hsel + 2v)
__device__ __forceinline__ int kA_map(int v, int hsel) {
  return (hsel << 4) + (v << 1);
}

__device__ __forceinline__ v8f wmma_bf16(const Frag& a, const Frag& b, v8f c) {
  return __builtin_amdgcn_wmma_f32_16x16x32_bf16(false, a.v, false, b.v,
                                                 (short)0, c, false, false);
}

// ---------------------------------------------------------------------------
// fp32 [K][N] -> bf16 [N][K] transpose-convert (weights, paid once per call)
// 32x32 LDS tile, grid = (N/32, K/32)
// ---------------------------------------------------------------------------
__global__ __launch_bounds__(256) void cvt_t_kernel(const float* __restrict__ w,
                                                    unsigned short* __restrict__ wT,
                                                    int K, int N) {
  __shared__ float t[32][33];
  const int k0 = blockIdx.y * 32, n0 = blockIdx.x * 32;
  const int tid = threadIdx.x;
  const int c = tid & 31, r8 = tid >> 5;
#pragma unroll
  for (int i = 0; i < 4; ++i) {
    int r = r8 + i * 8;
    t[r][c] = w[(size_t)(k0 + r) * N + n0 + c];
  }
  __syncthreads();
#pragma unroll
  for (int i = 0; i < 4; ++i) {
    int r = r8 + i * 8;
    wT[(size_t)(n0 + r) * K + k0 + c] = f2bf(t[c][r]);
  }
}

// ---------------------------------------------------------------------------
// LayerNorm (+ optional positional embedding) -> bf16 row
// one block per row, D = 1024, 256 threads x 4 elements
// ---------------------------------------------------------------------------
__global__ __launch_bounds__(256) void ln_kernel(const float* __restrict__ x,
                                                 const float* __restrict__ g,
                                                 const float* __restrict__ be,
                                                 const float* __restrict__ pos,
                                                 unsigned short* __restrict__ outp) {
  __shared__ float red[16];
  __shared__ float s_mu, s_rstd;
  const int row = blockIdx.x;
  const int tid = threadIdx.x;
  const float* xr = x + (size_t)row * D_;
  float vals[4];
  float lsum = 0.f, lsq = 0.f;
#pragma unroll
  for (int i = 0; i < 4; ++i) {
    float v = xr[tid + i * 256];
    vals[i] = v; lsum += v; lsq += v * v;
  }
#pragma unroll
  for (int msk = 1; msk < 32; msk <<= 1) {
    lsum += __shfl_xor(lsum, msk, 32);
    lsq  += __shfl_xor(lsq,  msk, 32);
  }
  int wid = tid >> 5, lane = tid & 31;
  if (lane == 0) { red[wid] = lsum; red[wid + 8] = lsq; }
  __syncthreads();
  if (tid == 0) {
    float ts = 0.f, tq = 0.f;
    for (int i = 0; i < 8; ++i) { ts += red[i]; tq += red[i + 8]; }
    float mu = ts / (float)D_;
    float var = tq / (float)D_ - mu * mu;
    s_mu = mu; s_rstd = rsqrtf(var + 1e-5f);
  }
  __syncthreads();
  float mu = s_mu, rstd = s_rstd;
  int srow = row % S_;
#pragma unroll
  for (int i = 0; i < 4; ++i) {
    int e = tid + i * 256;
    float v = (vals[i] - mu) * rstd * g[e] + be[e];
    if (pos) v += pos[(size_t)srow * D_ + e];
    outp[(size_t)row * D_ + e] = f2bf(v);
  }
}

// ---------------------------------------------------------------------------
// Tiled bf16 WMMA GEMM: C[M,N] = A[M,K] @ BT[N,K]^T + bias (+epilogue)
// B is pre-transposed to [N][K] so both tiles stage as contiguous dwords.
// Block tile 128x128, 8 waves (2M x 4N), wave tile 64x32 = 4x2 WMMA frags.
// Double-buffered LDS, register-staged global loads, one barrier / K-step.
// MODE 1: bias + residual -> f32 out
// MODE 2: bias + tanh-GELU -> bf16 out
// MODE 3: bias            -> bf16 out
// M%128==0, N%128==0, K%32==0 guaranteed by problem sizes.
// ---------------------------------------------------------------------------
template <int MODE>
__global__ __launch_bounds__(256) void gemm_wmma(const unsigned short* __restrict__ A,
                                                 const unsigned short* __restrict__ BT,
                                                 const float* __restrict__ bias,
                                                 const float* __restrict__ res,
                                                 void* __restrict__ Cout,
                                                 int M, int N, int K) {
  __shared__ __align__(16) unsigned short As[2][128 * 32];  // [m][k]
  __shared__ __align__(16) unsigned short Bs[2][128 * 32];  // [n][k]
  const int tid   = threadIdx.x;
  const int lane  = tid & 31, wid = tid >> 5;
  const int hsel  = lane >> 4, l16 = lane & 15;
  const int waveM = wid >> 2,  waveN = wid & 3;
  const int m0 = blockIdx.y * 128, n0 = blockIdx.x * 128;
  const int ldr = tid >> 4, ldc = (tid & 15) * 2;  // staging row / k-pair col

  v8f acc[4][2] = {};

  unsigned int ra[8], rb[8];
  // prologue: stage K-tile 0
#pragma unroll
  for (int i = 0; i < 8; ++i) {
    int r = ldr + i * 16;
    ra[i] = *(const unsigned int*)(A  + (size_t)(m0 + r) * K + ldc);
    rb[i] = *(const unsigned int*)(BT + (size_t)(n0 + r) * K + ldc);
  }
#pragma unroll
  for (int i = 0; i < 8; ++i) {
    int r = ldr + i * 16;
    *(unsigned int*)(As[0] + r * 32 + ldc) = ra[i];
    *(unsigned int*)(Bs[0] + r * 32 + ldc) = rb[i];
  }

  int buf = 0;
  for (int k0 = 0; k0 < K; k0 += 32) {
    __syncthreads();
    const bool has_next = (k0 + 32) < K;
    if (has_next) {
      // issue next tile's global loads (latency hidden behind WMMAs below)
#pragma unroll
      for (int i = 0; i < 8; ++i) {
        int r = ldr + i * 16;
        ra[i] = *(const unsigned int*)(A  + (size_t)(m0 + r) * K + k0 + 32 + ldc);
        rb[i] = *(const unsigned int*)(BT + (size_t)(n0 + r) * K + k0 + 32 + ldc);
      }
      if (k0 + 64 < K) {  // warm L2/WGP$ two tiles ahead (global_prefetch_b8)
        __builtin_prefetch(A  + (size_t)(m0 + (tid & 127)) * K + k0 + 64, 0, 1);
        __builtin_prefetch(BT + (size_t)(n0 + (tid & 127)) * K + k0 + 64, 0, 1);
      }
    }

    // fragments from current buffer
    Frag af[4], bfr[2];
#pragma unroll
    for (int mi = 0; mi < 4; ++mi) {
      int row = waveM * 64 + mi * 16 + l16;
#pragma unroll
      for (int v = 0; v < 8; ++v)
        af[mi].u[v] = *(const unsigned int*)(As[buf] + row * 32 + kA_map(v, hsel));
    }
#pragma unroll
    for (int ni = 0; ni < 2; ++ni) {
      int n = waveN * 32 + ni * 16 + l16;
#pragma unroll
      for (int v = 0; v < 8; ++v)
        bfr[ni].u[v] = *(const unsigned int*)(Bs[buf] + n * 32 + hsel * 16 + v * 2);
    }
#pragma unroll
    for (int mi = 0; mi < 4; ++mi)
#pragma unroll
      for (int ni = 0; ni < 2; ++ni)
        acc[mi][ni] = wmma_bf16(af[mi], bfr[ni], acc[mi][ni]);

    if (has_next) {
#pragma unroll
      for (int i = 0; i < 8; ++i) {
        int r = ldr + i * 16;
        *(unsigned int*)(As[buf ^ 1] + r * 32 + ldc) = ra[i];
        *(unsigned int*)(Bs[buf ^ 1] + r * 32 + ldc) = rb[i];
      }
    }
    buf ^= 1;
  }

  // epilogue
#pragma unroll
  for (int mi = 0; mi < 4; ++mi) {
#pragma unroll
    for (int ni = 0; ni < 2; ++ni) {
      int gcol = n0 + waveN * 32 + ni * 16 + l16;
      float bia = bias[gcol];
#pragma unroll
      for (int v = 0; v < 8; ++v) {
        int grow = m0 + waveM * 64 + mi * 16 + hsel * 8 + v;
        float val = acc[mi][ni][v] + bia;
        if (MODE == 1) {
          val += res[(size_t)grow * N + gcol];
          ((float*)Cout)[(size_t)grow * N + gcol] = val;
        } else if (MODE == 2) {
          float t = 0.7978845608028654f * (val + 0.044715f * val * val * val);
          val = 0.5f * val * (1.0f + tanhf(t));
          ((unsigned short*)Cout)[(size_t)grow * N + gcol] = f2bf(val);
        } else {
          ((unsigned short*)Cout)[(size_t)grow * N + gcol] = f2bf(val);
        }
      }
    }
  }
}

// ---------------------------------------------------------------------------
// Causal flash attention over bf16 qkv [B,S,3,H,HD].
// grid = (S/64 q-tiles, B*H). 128 threads = 4 waves; wave owns 16 q-rows.
// Key blocks of 64; only blocks 0..blockIdx.x are visited (causal).
// out: bf16 [B,S,H*HD] (already the transposed/reshaped layout).
// ---------------------------------------------------------------------------
__global__ __launch_bounds__(128) void attn_wmma(const unsigned short* __restrict__ qkv,
                                                 unsigned short* __restrict__ outp) {
  __shared__ __align__(16) unsigned short Ks[64 * 64];      // [key][d]
  __shared__ __align__(16) unsigned short Vs[64 * 64];      // [d][key] (transposed)
  __shared__ __align__(16) unsigned short Ps[4 * 16 * 64];  // per-wave P staging
  const int tid  = threadIdx.x;
  const int lane = tid & 31, wave = tid >> 5;
  const int hsel = lane >> 4, l16 = lane & 15;
  const int bh = blockIdx.y;
  const int b = bh / H_, h = bh % H_;
  const int qb = blockIdx.x * 64;

  // load this wave's Q fragments (A-matrix layout, d pairs contiguous)
  Frag qf0, qf1;
  {
    int s = qb + wave * 16 + l16;
    size_t base = ((size_t)(b * S_ + s) * 3 * H_ + h) * HD_;
#pragma unroll
    for (int v = 0; v < 8; ++v) {
      int ka = kA_map(v, hsel);
      qf0.u[v] = *(const unsigned int*)(qkv + base + ka);
      qf1.u[v] = *(const unsigned int*)(qkv + base + 32 + ka);
    }
  }

  float rm[8], rl[8];
#pragma unroll
  for (int v = 0; v < 8; ++v) { rm[v] = -1e30f; rl[v] = 0.f; }
  v8f o0 = {}, o1 = {}, o2 = {}, o3 = {};

  const int nkb = blockIdx.x + 1;
  for (int kb = 0; kb < nkb; ++kb) {
    const int kbase = kb * 64;
    // stage K tile [key][d] (direct copy, already B-frag friendly)
#pragma unroll
    for (int i = 0; i < 16; ++i) {
      int p = tid + i * 128;
      int r = p >> 5, cp = p & 31;
      size_t base = ((size_t)(b * S_ + kbase + r) * 3 * H_ + H_ + h) * HD_;
      *(unsigned int*)(Ks + r * 64 + cp * 2) =
          *(const unsigned int*)(qkv + base + cp * 2);
    }
    // stage V tile transposed [d][key]
#pragma unroll
    for (int i = 0; i < 16; ++i) {
      int p = tid + i * 128;
      int r = p >> 5, cp = p & 31;
      size_t base = ((size_t)(b * S_ + kbase + r) * 3 * H_ + 2 * H_ + h) * HD_;
      unsigned int val = *(const unsigned int*)(qkv + base + cp * 2);
      Vs[(cp * 2) * 64 + r]     = (unsigned short)(val & 0xFFFFu);
      Vs[(cp * 2 + 1) * 64 + r] = (unsigned short)(val >> 16);
    }
    __syncthreads();

    // scores S = Q @ K^T for 16 q-rows x 64 keys (4 n-tiles, K-dim = HD = 64)
    v8f sc[4];
#pragma unroll
    for (int ni = 0; ni < 4; ++ni) {
      Frag bf0, bf1;
#pragma unroll
      for (int v = 0; v < 8; ++v) {
        int kk = hsel * 16 + v * 2;
        bf0.u[v] = *(const unsigned int*)(Ks + (ni * 16 + l16) * 64 + kk);
        bf1.u[v] = *(const unsigned int*)(Ks + (ni * 16 + l16) * 64 + 32 + kk);
      }
      v8f s = {};
      s = wmma_bf16(qf0, bf0, s);
      s = wmma_bf16(qf1, bf1, s);
      sc[ni] = s;
    }

    // scale + causal mask, running row-max
    float mv[8];
#pragma unroll
    for (int v = 0; v < 8; ++v) mv[v] = -1e30f;
#pragma unroll
    for (int ni = 0; ni < 4; ++ni) {
      int col = kbase + ni * 16 + l16;
#pragma unroll
      for (int v = 0; v < 8; ++v) {
        int qi = qb + wave * 16 + hsel * 8 + v;
        float f = sc[ni][v] * 0.125f;   // 1/sqrt(64)
        if (col > qi) f = -1e30f;
        sc[ni][v] = f;
        mv[v] = fmaxf(mv[v], f);
      }
    }
#pragma unroll
    for (int v = 0; v < 8; ++v) {
#pragma unroll
      for (int msk = 1; msk < 16; msk <<= 1)
        mv[v] = fmaxf(mv[v], __shfl_xor(mv[v], msk, 32));
    }
    float alpha[8], psum[8];
#pragma unroll
    for (int v = 0; v < 8; ++v) {
      float mnew = fmaxf(rm[v], mv[v]);
      alpha[v] = __expf(rm[v] - mnew);
      rm[v] = mnew;
      psum[v] = 0.f;
    }
#pragma unroll
    for (int ni = 0; ni < 4; ++ni) {
#pragma unroll
      for (int v = 0; v < 8; ++v) {
        float p = __expf(sc[ni][v] - rm[v]);
        sc[ni][v] = p;
        psum[v] += p;
      }
    }
#pragma unroll
    for (int v = 0; v < 8; ++v) {
#pragma unroll
      for (int msk = 1; msk < 16; msk <<= 1)
        psum[v] += __shfl_xor(psum[v], msk, 32);
      rl[v] = rl[v] * alpha[v] + psum[v];
      o0[v] *= alpha[v]; o1[v] *= alpha[v]; o2[v] *= alpha[v]; o3[v] *= alpha[v];
    }

    // C-layout -> A-layout for P via per-wave LDS bounce (no barrier needed:
    // LDS ops are in-order within a wave)
#pragma unroll
    for (int ni = 0; ni < 4; ++ni) {
#pragma unroll
      for (int v = 0; v < 8; ++v)
        Ps[wave * 1024 + (hsel * 8 + v) * 64 + ni * 16 + l16] = f2bf(sc[ni][v]);
    }
    Frag pa0, pa1;
#pragma unroll
    for (int v = 0; v < 8; ++v) {
      int ka = kA_map(v, hsel);
      pa0.u[v] = *(const unsigned int*)(Ps + wave * 1024 + l16 * 64 + ka);
      pa1.u[v] = *(const unsigned int*)(Ps + wave * 1024 + l16 * 64 + 32 + ka);
    }
    // O += P @ V  (K-dim = 64 keys, 4 hd n-tiles)
#pragma unroll
    for (int f = 0; f < 4; ++f) {
      Frag vb0, vb1;
#pragma unroll
      for (int v = 0; v < 8; ++v) {
        int kk = hsel * 16 + v * 2;
        vb0.u[v] = *(const unsigned int*)(Vs + (f * 16 + l16) * 64 + kk);
        vb1.u[v] = *(const unsigned int*)(Vs + (f * 16 + l16) * 64 + 32 + kk);
      }
      v8f& o = (f == 0) ? o0 : (f == 1) ? o1 : (f == 2) ? o2 : o3;
      o = wmma_bf16(pa0, vb0, o);
      o = wmma_bf16(pa1, vb1, o);
    }
    __syncthreads();
  }

  // normalize and store to [B,S,H*HD] bf16
#pragma unroll
  for (int f = 0; f < 4; ++f) {
    v8f& o = (f == 0) ? o0 : (f == 1) ? o1 : (f == 2) ? o2 : o3;
#pragma unroll
    for (int v = 0; v < 8; ++v) {
      int s = qb + wave * 16 + hsel * 8 + v;
      int col = h * HD_ + f * 16 + l16;
      outp[(size_t)(b * S_ + s) * D_ + col] = f2bf(o[v] / rl[v]);
    }
  }
}

// ---------------------------------------------------------------------------
extern "C" void kernel_launch(void* const* d_in, const int* in_sizes, int n_in,
                              void* d_out, int out_size, void* d_ws, size_t ws_size,
                              hipStream_t stream) {
  (void)in_sizes; (void)n_in; (void)out_size; (void)ws_size;
  const float* x      = (const float*)d_in[0];
  const float* pos    = (const float*)d_in[1];
  const float* ln1_g  = (const float*)d_in[2];
  const float* ln1_b  = (const float*)d_in[3];
  const float* w_qkv  = (const float*)d_in[4];
  const float* b_qkv  = (const float*)d_in[5];
  const float* w_o    = (const float*)d_in[6];
  const float* b_o    = (const float*)d_in[7];
  const float* ln2_g  = (const float*)d_in[8];
  const float* ln2_b  = (const float*)d_in[9];
  const float* w_fc   = (const float*)d_in[10];
  const float* b_fc   = (const float*)d_in[11];
  const float* w_proj = (const float*)d_in[12];
  const float* b_proj = (const float*)d_in[13];

  char* ws = (char*)d_ws;
  size_t off = 0;
  auto alloc = [&](size_t bytes) -> void* {
    void* p = ws + off;
    off += (bytes + 255) & ~(size_t)255;
    return p;
  };
  unsigned short* h_bf     = (unsigned short*)alloc((size_t)MROWS * D_ * 2);
  unsigned short* qkv_bf   = (unsigned short*)alloc((size_t)MROWS * 3 * D_ * 2);
  unsigned short* attn_bf  = (unsigned short*)alloc((size_t)MROWS * D_ * 2);
  float*          y1       = (float*)alloc((size_t)MROWS * D_ * 4);
  unsigned short* m_bf     = (unsigned short*)alloc((size_t)MROWS * D_ * 2);
  unsigned short* a1_bf    = (unsigned short*)alloc((size_t)MROWS * DFF_ * 2);
  unsigned short* wqkvT    = (unsigned short*)alloc((size_t)D_ * 3 * D_ * 2);
  unsigned short* woT      = (unsigned short*)alloc((size_t)D_ * D_ * 2);
  unsigned short* wfcT     = (unsigned short*)alloc((size_t)D_ * DFF_ * 2);
  unsigned short* wprojT   = (unsigned short*)alloc((size_t)DFF_ * D_ * 2);

  // weights -> bf16, transposed to [N][K] (once per call; L2-resident after)
  cvt_t_kernel<<<dim3((3 * D_) / 32, D_ / 32), 256, 0, stream>>>(w_qkv, wqkvT, D_, 3 * D_);
  cvt_t_kernel<<<dim3(D_ / 32, D_ / 32), 256, 0, stream>>>(w_o, woT, D_, D_);
  cvt_t_kernel<<<dim3(DFF_ / 32, D_ / 32), 256, 0, stream>>>(w_fc, wfcT, D_, DFF_);
  cvt_t_kernel<<<dim3(D_ / 32, DFF_ / 32), 256, 0, stream>>>(w_proj, wprojT, DFF_, D_);

  // h = LN1(x) + pos_emb   (bf16)
  ln_kernel<<<MROWS, 256, 0, stream>>>(x, ln1_g, ln1_b, pos, h_bf);
  // qkv = h @ w_qkv + b_qkv   (bf16 out for attention)
  gemm_wmma<3><<<dim3((3 * D_) / 128, MROWS / 128), 256, 0, stream>>>(
      h_bf, wqkvT, b_qkv, nullptr, qkv_bf, MROWS, 3 * D_, D_);
  // causal flash attention
  attn_wmma<<<dim3(S_ / 64, B_ * H_), 128, 0, stream>>>(qkv_bf, attn_bf);
  // y1 = x + attn @ w_o + b_o   (f32)
  gemm_wmma<1><<<dim3(D_ / 128, MROWS / 128), 256, 0, stream>>>(
      attn_bf, woT, b_o, x, y1, MROWS, D_, D_);
  // m = LN2(y1)   (bf16)
  ln_kernel<<<MROWS, 256, 0, stream>>>(y1, ln2_g, ln2_b, nullptr, m_bf);
  // a1 = gelu(m @ w_fc + b_fc)   (bf16)
  gemm_wmma<2><<<dim3(DFF_ / 128, MROWS / 128), 256, 0, stream>>>(
      m_bf, wfcT, b_fc, nullptr, a1_bf, MROWS, DFF_, D_);
  // out = y1 + a1 @ w_proj + b_proj   (f32)
  gemm_wmma<1><<<dim3(D_ / 128, MROWS / 128), 256, 0, stream>>>(
      a1_bf, wprojT, b_proj, y1, (float*)d_out, MROWS, D_, DFF_);
}
